// HandmadeConv2dImplementation_52707838656978
// MI455X (gfx1250) — matrix operand order repaired and checked by reading, not compiled, not run on hardware
//
#include <hip/hip_runtime.h>

typedef __bf16 bf16;
typedef __attribute__((ext_vector_type(8)))  bf16  v8bf;
typedef __attribute__((ext_vector_type(16))) bf16  v16bf;
typedef __attribute__((ext_vector_type(2)))  float v2f;
typedef __attribute__((ext_vector_type(8)))  float v8f;

namespace {
constexpr int Cin  = 128;
constexpr int Hin  = 64;
constexpr int Win  = 64;
constexpr int Nimg = 16;
constexpr int Och  = 256;
constexpr int Pout = 62;
constexpr int Qout = 62;
constexpr int QT   = 4;                                      // 16-col q tiles
constexpr size_t SxElems = (size_t)Nimg * Hin * Win * Cin;   // real pixels/plane
constexpr size_t PadElems = 512;                             // zero pad/plane
constexpr size_t SxPad    = SxElems + PadElems;
constexpr size_t SwElems  = (size_t)9 * Och * Cin;           // per bf16 plane
}

__device__ __forceinline__ v16bf cat16(v8bf a, v8bf b) {
    return __builtin_shufflevector(a, b, 0, 1, 2, 3, 4, 5, 6, 7,
                                   8, 9, 10, 11, 12, 13, 14, 15);
}

// ---- zero the OOB pad region of both X planes (run every call) -------------
__global__ void zero_pad(bf16* __restrict__ Xh, bf16* __restrict__ Xl) {
    int t = threadIdx.x;                      // 256 threads
    bf16* ph = Xh + SxElems;
    bf16* pl = Xl + SxElems;
    ph[t]       = (bf16)0.0f;
    ph[t + 256] = (bf16)0.0f;
    pl[t]       = (bf16)0.0f;
    pl[t + 256] = (bf16)0.0f;
}

// ---- split x (NCHW f32) into hi/lo bf16 planes in NHWC ---------------------
__global__ __launch_bounds__(256) void x_split(const float* __restrict__ X,
                                               bf16* __restrict__ Xh,
                                               bf16* __restrict__ Xl) {
    int pix = blockIdx.x * blockDim.x + threadIdx.x;   // over Nimg*Hin*Win
    int w = pix & 63;
    int h = (pix >> 6) & 63;
    int n = pix >> 12;
    const float* src = X + (size_t)n * Cin * (Hin * Win) + h * Win + w;
    bf16* dh = Xh + (size_t)pix * Cin;
    bf16* dl = Xl + (size_t)pix * Cin;
#pragma unroll 2
    for (int c0 = 0; c0 < Cin; c0 += 8) {
        v8bf hv, lv;
#pragma unroll
        for (int j = 0; j < 8; ++j) {
            float v  = src[(size_t)(c0 + j) * (Hin * Win)];
            bf16  hi = (bf16)v;
            float r  = v - (float)hi;
            hv[j] = hi;
            lv[j] = (bf16)r;
        }
        *(v8bf*)(dh + c0) = hv;
        *(v8bf*)(dl + c0) = lv;
    }
}

// ---- split weights OIHW f32 into hi/lo bf16, layout [rs][c/32][o][32] ------
__global__ __launch_bounds__(256) void w_split(const float* __restrict__ W,
                                               bf16* __restrict__ Wh,
                                               bf16* __restrict__ Wl) {
    int idx = blockIdx.x * blockDim.x + threadIdx.x;   // over 9*4*256*32
    if (idx >= (int)SwElems) return;
    int ci = idx & 31;
    int o  = (idx >> 5) & 255;
    int cb = (idx >> 13) & 3;
    int rs = idx >> 15;
    int c  = cb * 32 + ci;
    float v  = W[((size_t)o * Cin + c) * 9 + rs];
    bf16  hi = (bf16)v;
    float r  = v - (float)hi;
    Wh[idx] = hi;
    Wl[idx] = (bf16)r;
}

// ---- main conv: bf16x3 split product on v_wmma_f32_16x16x32_bf16 -----------
__global__ __launch_bounds__(256) void conv_bf16x3(const bf16* __restrict__ Xh,
                                                   const bf16* __restrict__ Xl,
                                                   const bf16* __restrict__ Wh,
                                                   const bf16* __restrict__ Wl,
                                                   float* __restrict__ Out) {
    const int wave = threadIdx.x >> 5;
    const int lane = threadIdx.x & 31;
    const int half = lane >> 4;
    const int l16  = lane & 15;

    int b  = blockIdx.x;
    int qt = b % QT;   b /= QT;
    int p  = b % Pout; b /= Pout;
    int n  = b % Nimg; b /= Nimg;
    int mg = b;
    const int m_base = mg * 128 + wave * 16;
    const int q0 = qt * 16;

    const int  q      = q0 + l16;
    const bool qvalid = (q < Qout);

    // Per-r row base pointers; invalid q lanes read the zeroed pad region.
    // All later X offsets (s*128 + cb*32 + 16) stay inside the 512-elem pad.
    const bf16* xh_r[3];
    const bf16* xl_r[3];
#pragma unroll
    for (int r = 0; r < 3; ++r) {
        size_t off = qvalid
            ? ((((size_t)n * Hin + (p + r)) * Win + q0 + l16) * Cin)
            : SxElems;
        xh_r[r] = Xh + off + half * 8;
        xl_r[r] = Xl + off + half * 8;
    }

    // Per-lane weight base: lane = m, halves pick K {0-7,16-23}/{8-15,24-31}
    const bf16* whb = Wh + (size_t)(m_base + l16) * 32 + half * 8;
    const bf16* wlb = Wl + (size_t)(m_base + l16) * 32 + half * 8;

    v8f acc0 = {}, acc1 = {}, acc2 = {};

    for (int cb = 0; cb < 4; ++cb) {
#pragma unroll
        for (int rs = 0; rs < 9; ++rs) {          // fully unrolled: r,s const
            const int r = rs / 3;
            const int s = rs - 3 * r;
            const bf16* bh_p = xh_r[r] + (s * Cin + cb * 32);
            const bf16* bl_p = xl_r[r] + (s * Cin + cb * 32);
            const size_t wo  = (size_t)(rs * 4 + cb) * (Och * 32);

            v16bf ah = cat16(*(const v8bf*)(whb + wo),
                             *(const v8bf*)(whb + wo + 16));
            v16bf al = cat16(*(const v8bf*)(wlb + wo),
                             *(const v8bf*)(wlb + wo + 16));
            v16bf bh = cat16(*(const v8bf*)(bh_p),
                             *(const v8bf*)(bh_p + 16));
            v16bf bl = cat16(*(const v8bf*)(bl_p),
                             *(const v8bf*)(bl_p + 16));

            acc0 = __builtin_amdgcn_wmma_f32_16x16x32_bf16(
                false, ah, false, bh, (short)0, acc0, false, false);
            acc1 = __builtin_amdgcn_wmma_f32_16x16x32_bf16(
                false, ah, false, bl, (short)0, acc1, false, false);
            acc2 = __builtin_amdgcn_wmma_f32_16x16x32_bf16(
                false, al, false, bh, (short)0, acc2, false, false);
        }
    }

#pragma unroll
    for (int i = 0; i < 8; ++i) acc0[i] += acc1[i] + acc2[i];

    if (qvalid) {
        float* orow = Out + (((size_t)n * Och + m_base + half * 8) * Pout + p) * Qout + q;
#pragma unroll
        for (int i = 0; i < 8; ++i)
            orow[(size_t)i * (Pout * Qout)] = acc0[i];
    }
}

// ---- fallback: exact f32 WMMA straight from global (no workspace) ----------
__global__ __launch_bounds__(256) void conv_f32(const float* __restrict__ X,
                                                const float* __restrict__ W,
                                                float* __restrict__ Out) {
    const int wave = threadIdx.x >> 5;
    const int lane = threadIdx.x & 31;
    const int half = lane >> 4;
    const int l16  = lane & 15;

    int b  = blockIdx.x;
    int qt = b % QT;   b /= QT;
    int p  = b % Pout; b /= Pout;
    int n  = b % Nimg; b /= Nimg;
    int mg = b;
    const int m_base = mg * 128 + wave * 16;
    const int q0 = qt * 16;

    const int  q      = q0 + l16;
    const bool qvalid = (q < Qout);

    const float* xb = X + (size_t)n * Cin * (Hin * Win);
    v8f acc = {};
    for (int rs = 0; rs < 9; ++rs) {
        const int r = rs / 3;
        const int s = rs - 3 * r;
        const float* xrow = xb + (p + r) * Win + (q0 + s + l16);
        const float* xl   = qvalid ? xrow : xb;
        const float* wrs  = W + (size_t)(m_base + l16) * Cin * 9 + rs;
#pragma unroll 4
        for (int c0 = 0; c0 < Cin; c0 += 4) {
            const int c = c0 + 2 * half;
            v2f a, bf;
            a.x = wrs[(size_t)c * 9];
            a.y = wrs[(size_t)(c + 1) * 9];
            float bx = xl[(size_t)c * (Hin * Win)];
            float by = xl[(size_t)(c + 1) * (Hin * Win)];
            bf.x = qvalid ? bx : 0.0f;
            bf.y = qvalid ? by : 0.0f;
            acc = __builtin_amdgcn_wmma_f32_16x16x4_f32(
                false, a, false, bf, (short)0, acc, false, false);
        }
    }
    if (qvalid) {
        float* orow = Out + (((size_t)n * Och + m_base + half * 8) * Pout + p) * Qout + q;
#pragma unroll
        for (int i = 0; i < 8; ++i)
            orow[(size_t)i * (Pout * Qout)] = acc[i];
    }
}

extern "C" void kernel_launch(void* const* d_in, const int* in_sizes, int n_in,
                              void* d_out, int out_size, void* d_ws, size_t ws_size,
                              hipStream_t stream) {
    (void)in_sizes; (void)n_in; (void)out_size;
    const float* x = (const float*)d_in[0];
    const float* w = (const float*)d_in[1];
    float*       o = (float*)d_out;

    constexpr int    nblocks = 2 * Nimg * Pout * QT;             // 7936
    constexpr size_t need    = (2 * SxPad + 2 * SwElems) * sizeof(bf16);

    if (d_ws != nullptr && ws_size >= need) {
        bf16* Xh = (bf16*)d_ws;
        bf16* Xl = Xh + SxPad;
        bf16* Wh = Xl + SxPad;
        bf16* Wl = Wh + SwElems;
        zero_pad<<<1, 256, 0, stream>>>(Xh, Xl);
        x_split<<<(int)(SxElems / Cin) / 256, 256, 0, stream>>>(x, Xh, Xl);
        w_split<<<(int)((SwElems + 255) / 256), 256, 0, stream>>>(w, Wh, Wl);
        conv_bf16x3<<<nblocks, 256, 0, stream>>>(Xh, Xl, Wh, Wl, o);
    } else {
        conv_f32<<<nblocks, 256, 0, stream>>>(x, w, o);
    }
}
